// TriDirectionalMamba_22393959481641
// MI455X (gfx1250) — matrix operand order repaired and checked by reading, not compiled, not run on hardware
//
#include <hip/hip_runtime.h>

// ---------------------------------------------------------------------------
// Tri-directional Mamba block for gfx1250 (CDNA5).
// Pass 1: pre-swizzle f32 weights -> bf16 in WMMA B-fragment lane order (d_ws).
// Pass 2: per-pencil fused kernel: LN -> WMMA GEMM1 -> conv+SiLU -> WMMA GEMM2
//         -> fp32 selective scan -> gate -> WMMA GEMM3 -> atomic accumulate.
// B fragments: 2x global_load_b128 / lane. A fragments: 2x ds_load_b128 / lane.
// ---------------------------------------------------------------------------

typedef __bf16 bf16_t;
typedef bf16_t v16bf __attribute__((ext_vector_type(16)));
typedef float  v8f   __attribute__((ext_vector_type(8)));
typedef unsigned int v4u __attribute__((ext_vector_type(4)));

#define C_CH   96
#define DI     192
#define SEQ    32
#define NST    16
#define DTR    6
#define NDBL   38      // dt(6) + B(16) + C(16)
#define NDBLP  48      // padded to 3 N-tiles

// --- pre-swizzled weight arena in d_ws (bf16 elements) ---
// Win : 3 dirs x (3 kt x 24 nt) tiles x 512
// Wxp : 3 dirs x (6 kt x  3 nt) tiles x 512   (N padded 38 -> 48 with zeros)
// Wout: 3 dirs x (6 kt x  6 nt) tiles x 512
#define WIN_TPD   (3 * 24)          // tiles per dir
#define WXP_TPD   (6 * 3)
#define WOUT_TPD  (6 * 6)
#define OFF_WIN   0
#define OFF_WXP   (OFF_WIN + 3 * WIN_TPD * 512)    // 110592
#define OFF_WOUT  (OFF_WXP + 3 * WXP_TPD * 512)    // 138240
#define WS_ELEMS  (OFF_WOUT + 3 * WOUT_TPD * 512)  // 193536 (387 KB)

union FragU { v16bf v; v4u q[2]; unsigned short u[16]; };

static __device__ inline unsigned short f2bf(float f) {
  unsigned int u = __float_as_uint(f);
  u += 0x7FFFu + ((u >> 16) & 1u);          // round-to-nearest-even
  return (unsigned short)(u >> 16);
}
static __device__ inline float bf2f(unsigned short h) {
  return __uint_as_float(((unsigned int)h) << 16);
}
static __device__ inline float sigmoidf_(float x) { return 1.f / (1.f + __expf(-x)); }
static __device__ inline float softplusf_(float x) {
  return (x > 20.f) ? x : log1pf(__expf(x));
}

static __device__ inline v8f wmma_bf16(v16bf a, v16bf b, v8f c) {
  // (neg_a, A, neg_b, B, c_mod, C, reuse_a, reuse_b)
  return __builtin_amdgcn_wmma_f32_16x16x32_bf16(false, a, false, b, (short)0, c,
                                                 false, false);
}

// A fragment (16x32 bf16 tile) from LDS: two contiguous 8-element runs
// per lane (ISA 7.12.2) -> two ds_load_b128.
static __device__ inline v16bf load_a_frag(const unsigned short* lds, int row0,
                                           int stride, int k0, int lane) {
  FragU fu;
  const unsigned short* p =
      lds + (row0 + (lane & 15)) * stride + k0 + ((lane >> 4) << 3);
  fu.q[0] = *reinterpret_cast<const v4u*>(p);
  fu.q[1] = *reinterpret_cast<const v4u*>(p + 16);
  return fu.v;
}

// B fragment from the pre-swizzled bf16 arena: 32 contiguous bytes per lane
// -> two global_load_b128.
static __device__ inline v16bf load_b_pre(const unsigned short* base, int tile,
                                          int lane) {
  FragU fu;
  const v4u* p =
      reinterpret_cast<const v4u*>(base + ((size_t)tile * 32 + lane) * 16);
  fu.q[0] = p[0];
  fu.q[1] = p[1];
  return fu.v;
}

// ---------------------------------------------------------------------------
// Weight pre-swizzle: f32 row-major -> bf16 fragment-order arena.
// ---------------------------------------------------------------------------
__global__ __launch_bounds__(256) void prep_weights_kernel(
    const float* __restrict__ Win, const float* __restrict__ Wxp,
    const float* __restrict__ Wout, unsigned short* __restrict__ wsb) {
  for (int idx = blockIdx.x * 256 + threadIdx.x; idx < WS_ELEMS;
       idx += gridDim.x * 256) {
    int e = idx, dir, kt, nt, lane, i, k, n;
    float v;
    if (e < OFF_WXP) {  // Win: 96 x 384
      dir = e / (WIN_TPD * 512); int r = e % (WIN_TPD * 512);
      int tile = r / 512, s = r % 512;
      lane = s >> 4; i = s & 15;
      kt = tile / 24; nt = tile % 24;
      k = kt * 32 + ((lane >> 4) << 3) + (i & 7) + ((i >> 3) << 4);
      n = nt * 16 + (lane & 15);
      v = Win[(size_t)dir * C_CH * (2 * DI) + (size_t)k * (2 * DI) + n];
    } else if (e < OFF_WOUT) {  // Wxp: 192 x 38 (padded to 48)
      e -= OFF_WXP;
      dir = e / (WXP_TPD * 512); int r = e % (WXP_TPD * 512);
      int tile = r / 512, s = r % 512;
      lane = s >> 4; i = s & 15;
      kt = tile / 3; nt = tile % 3;
      k = kt * 32 + ((lane >> 4) << 3) + (i & 7) + ((i >> 3) << 4);
      n = nt * 16 + (lane & 15);
      v = (n < NDBL) ? Wxp[(size_t)dir * DI * NDBL + (size_t)k * NDBL + n] : 0.f;
    } else {  // Wout: 192 x 96
      e -= OFF_WOUT;
      dir = e / (WOUT_TPD * 512); int r = e % (WOUT_TPD * 512);
      int tile = r / 512, s = r % 512;
      lane = s >> 4; i = s & 15;
      kt = tile / 6; nt = tile % 6;
      k = kt * 32 + ((lane >> 4) << 3) + (i & 7) + ((i >> 3) << 4);
      n = nt * 16 + (lane & 15);
      v = Wout[(size_t)dir * DI * C_CH + (size_t)k * C_CH + n];
    }
    wsb[idx] = f2bf(v);
  }
}

// --------------------------- LDS layout (57.6 KB) ---------------------------
// [     0 .. 24576) xmf   f32  [32][192]   (P1: xraw f32 [32][96])
// [ 24576 .. 36864) resb  bf16 [32][192]
// [ 36864 .. 49152) xmb   bf16 [32][192]   (later reused as yb)
// [ 49152 .. 55296) xlnb  bf16 [32][96]    (later reused as xdbl f32 [32][48])
// [ 55296 .. 57600) scr   f32  [576]
#define SM_BYTES 57600

__global__ __launch_bounds__(256, 1) void tri_mamba_kernel(
    const float* __restrict__ x, const float* __restrict__ ln_g,
    const float* __restrict__ ln_b, const float* __restrict__ Wconv,
    const float* __restrict__ bconv, const float* __restrict__ Wdt,
    const float* __restrict__ bdt, const float* __restrict__ A_log,
    const float* __restrict__ Dskip, const float* __restrict__ alpha,
    const unsigned short* __restrict__ wsb, float* __restrict__ out) {
  __shared__ __align__(16) unsigned char smem[SM_BYTES];
  float*          xmf  = (float*)(smem);
  unsigned short* resb = (unsigned short*)(smem + 24576);
  unsigned short* xmb  = (unsigned short*)(smem + 36864);
  unsigned short* xlnb = (unsigned short*)(smem + 49152);
  float*          xdbl = (float*)(smem + 49152);
  float*          scr  = (float*)(smem + 55296);

  const int sid = blockIdx.x;       // pencil id within direction
  const int dir = blockIdx.y;       // 0: along D, 1: along H, 2: along W
  const int tid = threadIdx.x;
  const int lane = tid & 31;
  const int wid = tid >> 5;

  // token t of this pencil sits at flat index c*32768 + soff + t*tstr
  int soff, tstr;
  if (dir == 0)      { soff = ((sid >> 5) << 5) + (sid & 31);  tstr = 1024; }
  else if (dir == 1) { soff = ((sid >> 5) << 10) + (sid & 31); tstr = 32;   }
  else               { soff = ((sid >> 5) << 10) + ((sid & 31) << 5); tstr = 1; }

  const unsigned short* WinB  = wsb + OFF_WIN  + (size_t)dir * WIN_TPD * 512;
  const unsigned short* WxpB  = wsb + OFF_WXP  + (size_t)dir * WXP_TPD * 512;
  const unsigned short* WoutB = wsb + OFF_WOUT + (size_t)dir * WOUT_TPD * 512;
  const float* Wconv_d = Wconv + (size_t)dir * DI * 4;
  const float* bconv_d = bconv + (size_t)dir * DI;
  const float* Wdt_d   = Wdt   + (size_t)dir * DTR * DI;
  const float* bdt_d   = bdt   + (size_t)dir * DI;
  const float* Alog_d  = A_log + (size_t)dir * DI * NST;
  const float* Dsk_d   = Dskip + (size_t)dir * DI;

  // ---------------- P1: load pencil + LayerNorm -> xlnb (bf16) --------------
  {
    const int t = tid >> 3, j = tid & 7;  // 32 tokens x 8 threads
    float s1 = 0.f, s2 = 0.f;
#pragma unroll
    for (int cc = 0; cc < 12; ++cc) {
      int c = j * 12 + cc;
      float v = x[(size_t)c * 32768 + soff + t * tstr];
      xmf[t * C_CH + c] = v;
      s1 += v; s2 += v * v;
    }
    scr[tid] = s1; scr[256 + tid] = s2;
    __syncthreads();
    if (j == 0) {
      float a = 0.f, b = 0.f;
#pragma unroll
      for (int k = 0; k < 8; ++k) { a += scr[t * 8 + k]; b += scr[256 + t * 8 + k]; }
      float mean = a * (1.f / C_CH);
      float var = b * (1.f / C_CH) - mean * mean;
      scr[512 + t * 2] = mean;
      scr[513 + t * 2] = rsqrtf(var + 1e-5f);
    }
    __syncthreads();
    float mean = scr[512 + t * 2], rstd = scr[513 + t * 2];
#pragma unroll
    for (int cc = 0; cc < 12; ++cc) {
      int c = j * 12 + cc;
      float g = ln_g[dir * C_CH + c], bb = ln_b[dir * C_CH + c];
      float v = (xmf[t * C_CH + c] - mean) * rstd * g + bb;
      xlnb[t * C_CH + c] = f2bf(v);
    }
  }
  __syncthreads();

  // ---------------- P2: GEMM1  xr[32][384] = xln @ Win ----------------------
  {
    v8f acc[3][2];
#pragma unroll
    for (int q = 0; q < 3; ++q) { acc[q][0] = {}; acc[q][1] = {}; }
#pragma unroll
    for (int kt = 0; kt < 3; ++kt) {
      v16bf a0 = load_a_frag(xlnb, 0, C_CH, kt * 32, lane);
      v16bf a1 = load_a_frag(xlnb, 16, C_CH, kt * 32, lane);
#pragma unroll
      for (int q = 0; q < 3; ++q) {
        v16bf bf = load_b_pre(WinB, kt * 24 + (wid * 3 + q), lane);
        acc[q][0] = wmma_bf16(a0, bf, acc[q][0]);
        acc[q][1] = wmma_bf16(a1, bf, acc[q][1]);
      }
    }
#pragma unroll
    for (int q = 0; q < 3; ++q)
#pragma unroll
      for (int m = 0; m < 2; ++m)
#pragma unroll
        for (int r = 0; r < 8; ++r) {
          int row = m * 16 + r + ((lane >> 4) << 3);
          int col = (wid * 3 + q) * 16 + (lane & 15);
          float v = acc[q][m][r];
          if (col < DI) xmf[row * DI + col] = v;          // wave-uniform branch
          else          resb[row * DI + (col - DI)] = f2bf(v);
        }
  }
  __syncthreads();

  // ---------------- P3: depthwise causal conv(4) + SiLU ---------------------
  {
    float r[24];
#pragma unroll
    for (int i = 0; i < 24; ++i) {
      int idx = i * 256 + tid;
      int t = idx / DI, d = idx % DI;
      float acc = bconv_d[d];
#pragma unroll
      for (int j = 0; j < 4; ++j) {
        int tt = t - 3 + j;
        if (tt >= 0) acc += Wconv_d[d * 4 + j] * xmf[tt * DI + d];
      }
      r[i] = acc;
    }
    __syncthreads();
#pragma unroll
    for (int i = 0; i < 24; ++i) {
      int idx = i * 256 + tid;
      int t = idx / DI, d = idx % DI;
      float sv = r[i] * sigmoidf_(r[i]);
      xmf[t * DI + d] = sv;           // u for the scan (f32)
      xmb[t * DI + d] = f2bf(sv);     // A-matrix for GEMM2 (bf16)
    }
  }
  __syncthreads();

  // ---------------- P4: GEMM2  x_dbl[32][38] = xm @ Wxp ---------------------
  if (wid < 6) {
    int m = wid & 1, nt = wid >> 1;
    v8f acc = {};
#pragma unroll
    for (int kt = 0; kt < 6; ++kt) {
      v16bf a = load_a_frag(xmb, m * 16, DI, kt * 32, lane);
      v16bf b = load_b_pre(WxpB, kt * 3 + nt, lane);
      acc = wmma_bf16(a, b, acc);
    }
#pragma unroll
    for (int r = 0; r < 8; ++r) {
      int row = m * 16 + r + ((lane >> 4) << 3);
      int col = nt * 16 + (lane & 15);
      xdbl[row * NDBLP + col] = acc[r];
    }
  }
  __syncthreads();

  // ---------------- P5: selective scan (fp32 recurrence) --------------------
  if (tid < DI) {
    const int d = tid;
    float Ar[NST], s[NST];
#pragma unroll
    for (int n = 0; n < NST; ++n) { Ar[n] = -__expf(Alog_d[d * NST + n]); s[n] = 0.f; }
    float wdt[DTR];
#pragma unroll
    for (int j = 0; j < DTR; ++j) wdt[j] = Wdt_d[j * DI + d];
    const float bdtd = bdt_d[d], dsk = Dsk_d[d];
    for (int t = 0; t < SEQ; ++t) {
      const float* xd = xdbl + t * NDBLP;
      float dt = bdtd;
#pragma unroll
      for (int j = 0; j < DTR; ++j) dt += xd[j] * wdt[j];
      float delta = softplusf_(dt);
      float u = xmf[t * DI + d];
      float du = delta * u;
      float y = 0.f;
#pragma unroll
      for (int n = 0; n < NST; ++n) {
        float dA = __expf(delta * Ar[n]);
        s[n] = dA * s[n] + du * xd[DTR + n];
        y += s[n] * xd[DTR + NST + n];
      }
      y += u * dsk;
      float rv = bf2f(resb[t * DI + d]);
      y *= rv * sigmoidf_(rv);        // gate with silu(res)
      xmb[t * DI + d] = f2bf(y);      // yb reuses the xmb region
    }
  }
  __syncthreads();

  // ---------------- P6: GEMM3  out[32][96] = y @ Wout, scaled accumulate ----
  if (wid < 6) {
    v8f acc0 = {}, acc1 = {};
#pragma unroll
    for (int kt = 0; kt < 6; ++kt) {
      v16bf a0 = load_a_frag(xmb, 0, DI, kt * 32, lane);
      v16bf a1 = load_a_frag(xmb, 16, DI, kt * 32, lane);
      v16bf b = load_b_pre(WoutB, kt * 6 + wid, lane);
      acc0 = wmma_bf16(a0, b, acc0);
      acc1 = wmma_bf16(a1, b, acc1);
    }
    float a0v = alpha[0], a1v = alpha[1], a2v = alpha[2];
    float mx = fmaxf(a0v, fmaxf(a1v, a2v));
    float e0 = __expf(a0v - mx), e1 = __expf(a1v - mx), e2 = __expf(a2v - mx);
    float wdir = ((dir == 0) ? e0 : (dir == 1) ? e1 : e2) / (e0 + e1 + e2);

#pragma unroll
    for (int m = 0; m < 2; ++m)
#pragma unroll
      for (int r = 0; r < 8; ++r) {
        int t = m * 16 + r + ((lane >> 4) << 3);
        int c = wid * 16 + (lane & 15);
        float v = ((m == 0) ? acc0[r] : acc1[r]) * wdir;
        atomicAdd(&out[(size_t)c * 32768 + soff + t * tstr], v);
      }
  }
}

extern "C" void kernel_launch(void* const* d_in, const int* in_sizes, int n_in,
                              void* d_out, int out_size, void* d_ws, size_t ws_size,
                              hipStream_t stream) {
  (void)in_sizes; (void)n_in; (void)ws_size;
  const float* x     = (const float*)d_in[0];
  const float* ln_g  = (const float*)d_in[1];
  const float* ln_b  = (const float*)d_in[2];
  const float* Win   = (const float*)d_in[3];
  const float* Wconv = (const float*)d_in[4];
  const float* bconv = (const float*)d_in[5];
  const float* Wxp   = (const float*)d_in[6];
  const float* Wdt   = (const float*)d_in[7];
  const float* bdt   = (const float*)d_in[8];
  const float* A_log = (const float*)d_in[9];
  const float* Dskip = (const float*)d_in[10];
  const float* Wout  = (const float*)d_in[11];
  const float* alpha = (const float*)d_in[12];
  float* out = (float*)d_out;
  unsigned short* wsb = (unsigned short*)d_ws;

  hipMemsetAsync(out, 0, (size_t)out_size * sizeof(float), stream);
  prep_weights_kernel<<<(WS_ELEMS + 255) / 256, 256, 0, stream>>>(Win, Wxp, Wout,
                                                                  wsb);
  dim3 grid(1024, 3, 1);
  tri_mamba_kernel<<<grid, 256, 0, stream>>>(x, ln_g, ln_b, Wconv, bconv, Wdt,
                                             bdt, A_log, Dskip, alpha, wsb, out);
}
